// Attention_87729001988230
// MI455X (gfx1250) — matrix operand order repaired and checked by reading, not compiled, not run on hardware
//
#include <hip/hip_runtime.h>
#include <hip/hip_bf16.h>

#define H 512
#define BB 8
#define QQ 2048
#define KK 2048
#define KC 128

typedef __attribute__((ext_vector_type(16))) __bf16 v16bf;
typedef __attribute__((ext_vector_type(8)))  float  v8f;

static __device__ __forceinline__ v8f wmma_bf16(v16bf a, v16bf b, v8f c) {
    return __builtin_amdgcn_wmma_f32_16x16x32_bf16(false, a, false, b, (short)0, c, false, false);
}

// ---------------------------------------------------------------------------
// Kernel 0: f32 -> bf16 bulk convert (used once for Wq / Wk; L2-resident after)
// ---------------------------------------------------------------------------
__global__ __launch_bounds__(256) void cvt_bf16_kernel(
    const float* __restrict__ src, __bf16* __restrict__ dst, int n)
{
    int i = (blockIdx.x * 256 + threadIdx.x) * 8;
    if (i + 8 <= n) {
        float4 a = *(const float4*)(src + i);
        float4 b = *(const float4*)(src + i + 4);
        union { __bf16 o[8]; uint4 u; } u;
        u.o[0] = (__bf16)a.x; u.o[1] = (__bf16)a.y; u.o[2] = (__bf16)a.z; u.o[3] = (__bf16)a.w;
        u.o[4] = (__bf16)b.x; u.o[5] = (__bf16)b.y; u.o[6] = (__bf16)b.z; u.o[7] = (__bf16)b.w;
        *(uint4*)(dst + i) = u.u;
    }
}

// ---------------------------------------------------------------------------
// Kernel 1: out_row = LayerNorm(x @ W^T + bias) * gamma + beta, bf16 output in
// [B][N][H] (batch-major) layout. One block = 32 rows; A tile staged once in
// LDS as bf16; W pre-converted to bf16; each wave owns 64 output columns and
// computes 2 row-tiles x 4 col-tiles of WMMA, reusing every B fragment twice.
// LayerNorm reduced straight from accumulator registers (no f32 LDS staging).
// ---------------------------------------------------------------------------
__global__ __launch_bounds__(256) void proj_ln_kernel(
    const float* __restrict__ X,      // [N*B, H]
    const __bf16* __restrict__ Wb,    // [H, H] bf16
    const float* __restrict__ bias,
    const float* __restrict__ gamma,
    const float* __restrict__ beta,
    __bf16* __restrict__ Out,         // [B][N][H]
    int N)                            // rows per batch (Q or K)
{
    __shared__ __align__(16) __bf16 as[32][H + 8];
    __shared__ float psum[32][8];
    __shared__ float psq[32][8];
    __shared__ float stat[32][2];

    const int tid  = threadIdx.x;
    const int wave = tid >> 5;
    const int lane = tid & 31;
    const int hi   = lane >> 4;
    const int ln   = lane & 15;
    const long row0 = (long)blockIdx.x * 32;

    // Stage + convert the 32x512 A tile once (f32 -> bf16 into padded LDS)
    for (int i = tid; i < 32 * (H / 4); i += 256) {
        int r = i / (H / 4), c = (i % (H / 4)) * 4;
        float4 v = *(const float4*)(X + (row0 + r) * H + c);
        __bf16* d = &as[r][c];
        d[0] = (__bf16)v.x; d[1] = (__bf16)v.y; d[2] = (__bf16)v.z; d[3] = (__bf16)v.w;
    }
    __syncthreads();

    v8f acc[2][4];
    #pragma unroll
    for (int rt = 0; rt < 2; ++rt)
        #pragma unroll
        for (int t = 0; t < 4; ++t) acc[rt][t] = (v8f){0,0,0,0,0,0,0,0};

    for (int k0 = 0; k0 < H; k0 += 32) {
        v16bf a0, a1;
        #pragma unroll
        for (int e = 0; e < 16; ++e) {
            int k = k0 + e + hi * 16;   // contiguous k = 16*hi + e (matches B)
            a0[e] = as[ln][k];
            a1[e] = as[16 + ln][k];
        }
        #pragma unroll
        for (int t = 0; t < 4; ++t) {
            int n = wave * 64 + t * 16 + ln;                // column = row of W
            const __bf16* wr = Wb + (long)n * H + k0 + hi * 16;
            v16bf bm;
            #pragma unroll
            for (int e = 0; e < 16; ++e) bm[e] = wr[e];
            acc[0][t] = wmma_bf16(a0, bm, acc[0][t]);
            acc[1][t] = wmma_bf16(a1, bm, acc[1][t]);
        }
    }

    // Add bias; per-row partial sum / sumsq via shuffles -> LDS partials
    #pragma unroll
    for (int rt = 0; rt < 2; ++rt) {
        #pragma unroll
        for (int g = 0; g < 8; ++g) {
            float s = 0.f, q = 0.f;
            #pragma unroll
            for (int t = 0; t < 4; ++t) {
                float bv = bias[wave * 64 + t * 16 + ln];
                float v = acc[rt][t][g] + bv;
                acc[rt][t][g] = v;
                s += v; q += v * v;
            }
            #pragma unroll
            for (int o = 8; o > 0; o >>= 1) {
                s += __shfl_xor(s, o, 32);
                q += __shfl_xor(q, o, 32);
            }
            if (ln == 0) {
                int row = rt * 16 + g + hi * 8;
                psum[row][wave] = s;
                psq[row][wave]  = q;
            }
        }
    }
    __syncthreads();
    if (tid < 32) {
        float s = 0.f, q = 0.f;
        #pragma unroll
        for (int j = 0; j < 8; ++j) { s += psum[tid][j]; q += psq[tid][j]; }
        float mu  = s * (1.0f / H);
        float var = q * (1.0f / H) - mu * mu;
        stat[tid][0] = mu;
        stat[tid][1] = rsqrtf(var + 1e-5f);
    }
    __syncthreads();

    // Normalize directly from registers and store bf16 to [B][N][H]
    #pragma unroll
    for (int rt = 0; rt < 2; ++rt) {
        #pragma unroll
        for (int t = 0; t < 4; ++t) {
            int n = wave * 64 + t * 16 + ln;
            float ga = gamma[n], be = beta[n];
            #pragma unroll
            for (int g = 0; g < 8; ++g) {
                int row = rt * 16 + g + hi * 8;
                long grow = row0 + row;
                long nidx = grow / BB;
                long bidx = grow % BB;
                float v = (acc[rt][t][g] - stat[row][0]) * stat[row][1] * ga + be;
                Out[(bidx * N + nidx) * H + n] = (__bf16)v;
            }
        }
    }
}

// ---------------------------------------------------------------------------
// Kernel 2: value [K,B,H] f32  ->  VT [B,H,K] bf16  (LDS-tiled transpose)
// ---------------------------------------------------------------------------
__global__ __launch_bounds__(256) void transpose_v_kernel(
    const float* __restrict__ V,
    __bf16* __restrict__ VT)
{
    __shared__ float tile[64][65];
    const int k0 = blockIdx.x * 64;
    const int h0 = blockIdx.y * 64;
    const int b  = blockIdx.z;
    const int tid = threadIdx.x;

    #pragma unroll
    for (int i = 0; i < 16; ++i) {
        int idx = tid + i * 256;
        int kl = idx >> 6, hl = idx & 63;
        tile[kl][hl] = V[((long)(k0 + kl) * BB + b) * H + h0 + hl];
    }
    __syncthreads();
    #pragma unroll
    for (int i = 0; i < 16; ++i) {
        int idx = tid + i * 256;
        int hl = idx >> 6, kl = idx & 63;
        VT[((long)b * H + h0 + hl) * KK + k0 + kl] = (__bf16)tile[kl][hl];
    }
}

// ---------------------------------------------------------------------------
// Kernel 3: flash-style masked attention, 32 query rows per block.
// Each wave: one 16-col K slice -> 2 score tiles per 128-wide chunk (K
// fragments reused for both row tiles), then accumulates its own 64-column
// H slice of the context (2x4 C tiles, V fragments reused twice).
// q tile staged in LDS via gfx1250 async-to-LDS DMA (ASYNCcnt tracked).
// ---------------------------------------------------------------------------
__global__ __launch_bounds__(256) void attn_kernel(
    const __bf16* __restrict__ Qb,   // [B][Q][H]
    const __bf16* __restrict__ Kb,   // [B][K][H]
    const __bf16* __restrict__ VT,   // [B][H][K]
    const unsigned char* __restrict__ qmask,  // [Q][B] bool
    const unsigned char* __restrict__ kmask,  // [K][B] bool
    float* __restrict__ Out)         // [Q][B][H]
{
    __shared__ __align__(16) __bf16 qs[32][H + 8];
    __shared__ __align__(16) __bf16 P[32][KC + 8];
    __shared__ float cmax[32][8];
    __shared__ float csum[32][8];
    __shared__ float mrow[32], lrow[32], scl[32], qm_s[32];

    const int tid  = threadIdx.x;
    const int wave = tid >> 5;
    const int lane = tid & 31;
    const int hi   = lane >> 4;
    const int ln   = lane & 15;
    const int b    = blockIdx.y;
    const long q0  = (long)blockIdx.x * 32;
    const float NEG_INF = -__builtin_inff();

    // ---- async DMA the 32x512 bf16 q tile into padded LDS (16B per lane)
    {
        unsigned lds_base = (unsigned)(unsigned long long)&qs[0][0];
        const __bf16* gbase = Qb + ((long)b * QQ + q0) * H;
        for (int i = tid; i < 32 * 64; i += 256) {      // 64 x 16B chunks per row
            int r = i >> 6, cb = (i & 63) * 16;          // byte offset in row
            unsigned laddr = lds_base + (unsigned)(r * ((H + 8) * 2) + cb);
            unsigned long long gaddr =
                (unsigned long long)(const void*)(gbase + (long)r * H + cb / 2);
            asm volatile("global_load_async_to_lds_b128 %0, %1, off"
                         :: "v"(laddr), "v"(gaddr) : "memory");
        }
        asm volatile("s_wait_asynccnt 0" ::: "memory");
    }
    if (tid < 32) {
        qm_s[tid] = qmask[(q0 + tid) * BB + b] ? 1.f : 0.f;
        mrow[tid] = NEG_INF;
        lrow[tid] = 0.f;
    }
    __syncthreads();

    v8f acc[2][4];
    #pragma unroll
    for (int rt = 0; rt < 2; ++rt)
        #pragma unroll
        for (int t = 0; t < 4; ++t) acc[rt][t] = (v8f){0,0,0,0,0,0,0,0};

    const float inv_sqrt_h = 0.04419417382415922f;  // 1/sqrt(512)

    for (int kc = 0; kc < KK; kc += KC) {
        const int col = kc + wave * 16 + ln;   // this lane's key column

        if (kc + KC < KK) {
            __builtin_prefetch(Kb + ((long)b * KK + col + KC) * H, 0, 0);
            __builtin_prefetch(VT + ((long)b * H + wave * 64 + ln) * KK + kc + KC, 0, 0);
        }

        // ---- scores: two 16x16 tiles (rows 0-15, 16-31) vs this wave's cols
        v8f sc0 = (v8f){0,0,0,0,0,0,0,0};
        v8f sc1 = (v8f){0,0,0,0,0,0,0,0};
        #pragma unroll 4
        for (int h0 = 0; h0 < H; h0 += 32) {
            v16bf a0, a1, bm;
            #pragma unroll
            for (int e = 0; e < 16; ++e) {
                int k = h0 + e + hi * 16;
                a0[e] = qs[ln][k];
                a1[e] = qs[16 + ln][k];
            }
            const __bf16* krow = Kb + ((long)b * KK + col) * H + h0 + hi * 16;
            #pragma unroll
            for (int e = 0; e < 16; ++e) bm[e] = krow[e];
            sc0 = wmma_bf16(a0, bm, sc0);
            sc1 = wmma_bf16(a1, bm, sc1);
        }

        const float kmv = kmask[(long)col * BB + b] ? 1.f : 0.f;
        float sv[2][8];
        #pragma unroll
        for (int rt = 0; rt < 2; ++rt) {
            #pragma unroll
            for (int g = 0; g < 8; ++g) {
                int row = rt * 16 + g + hi * 8;
                float s = (rt ? sc1[g] : sc0[g]) * inv_sqrt_h;
                bool msk = (kmv == 0.f) || (qm_s[row] == 0.f);
                sv[rt][g] = msk ? NEG_INF : s;
                float r = sv[rt][g];
                #pragma unroll
                for (int o = 8; o > 0; o >>= 1)
                    r = fmaxf(r, __shfl_xor(r, o, 32));
                if (ln == 0) cmax[row][wave] = r;
            }
        }
        __syncthreads();

        // ---- online max / rescale factors (per row)
        if (tid < 32) {
            float mo = mrow[tid], cm = mo;
            #pragma unroll
            for (int w2 = 0; w2 < 8; ++w2) cm = fmaxf(cm, cmax[tid][w2]);
            float f = (cm == NEG_INF) ? 1.f : __expf(2.f * (mo - cm));
            scl[tid]  = f;
            mrow[tid] = cm;
            lrow[tid] *= f;
        }
        __syncthreads();

        // ---- P = exp(s - 2*m), chunk row-sums, rescale accumulators
        #pragma unroll
        for (int rt = 0; rt < 2; ++rt) {
            #pragma unroll
            for (int g = 0; g < 8; ++g) {
                int row = rt * 16 + g + hi * 8;
                float mn = mrow[row];
                float p = (sv[rt][g] == NEG_INF) ? 0.f : __expf(sv[rt][g] - 2.f * mn);
                P[row][wave * 16 + ln] = (__bf16)p;
                float r = p;
                #pragma unroll
                for (int o = 8; o > 0; o >>= 1) r += __shfl_xor(r, o, 32);
                if (ln == 0) csum[row][wave] = r;
                float f = scl[row];
                #pragma unroll
                for (int t = 0; t < 4; ++t) acc[rt][t][g] *= f;
            }
        }
        __syncthreads();

        if (tid < 32) {
            float s = 0.f;
            #pragma unroll
            for (int w2 = 0; w2 < 8; ++w2) s += csum[tid][w2];
            lrow[tid] += s;
        }

        // ---- context: acc += P(32x128) . V(128 x this wave's 64 cols)
        #pragma unroll
        for (int kk = 0; kk < KC; kk += 32) {
            v16bf a0, a1;
            #pragma unroll
            for (int e = 0; e < 16; ++e) {
                int k = kk + e + hi * 16;
                a0[e] = P[ln][k];
                a1[e] = P[16 + ln][k];
            }
            #pragma unroll
            for (int t = 0; t < 4; ++t) {
                int h = wave * 64 + t * 16 + ln;
                const __bf16* vrow = VT + ((long)b * H + h) * KK + kc + kk + hi * 16;
                v16bf bm;
                #pragma unroll
                for (int e = 0; e < 16; ++e) bm[e] = vrow[e];
                acc[0][t] = wmma_bf16(a0, bm, acc[0][t]);
                acc[1][t] = wmma_bf16(a1, bm, acc[1][t]);
            }
        }
        __syncthreads();
    }

    // ---- epilogue: out = acc / (l + 0.001), scatter to [Q,B,H] f32
    #pragma unroll
    for (int rt = 0; rt < 2; ++rt) {
        #pragma unroll
        for (int t = 0; t < 4; ++t) {
            int h = wave * 64 + t * 16 + ln;
            #pragma unroll
            for (int g = 0; g < 8; ++g) {
                int row = rt * 16 + g + hi * 8;
                float denom = lrow[row] + 0.001f;
                Out[((q0 + row) * BB + b) * H + h] = acc[rt][t][g] / denom;
            }
        }
    }
}

// ---------------------------------------------------------------------------
extern "C" void kernel_launch(void* const* d_in, const int* in_sizes, int n_in,
                              void* d_out, int out_size, void* d_ws, size_t ws_size,
                              hipStream_t stream) {
    const float* query  = (const float*)d_in[0];
    const float* key_in = (const float*)d_in[1];
    const float* value  = (const float*)d_in[2];
    const unsigned char* qmask = (const unsigned char*)d_in[3];
    const unsigned char* kmask = (const unsigned char*)d_in[4];
    const float* Wq    = (const float*)d_in[5];
    const float* bq    = (const float*)d_in[6];
    const float* gq    = (const float*)d_in[7];
    const float* betaq = (const float*)d_in[8];
    const float* Wk    = (const float*)d_in[9];
    const float* bk    = (const float*)d_in[10];
    const float* gk    = (const float*)d_in[11];
    const float* betak = (const float*)d_in[12];

    char* ws = (char*)d_ws;
    const size_t tensor_bytes = (size_t)BB * QQ * H * 2;  // 16 MB each
    const size_t w_bytes = (size_t)H * H * 2;             // 512 KB each
    __bf16* qb  = (__bf16*)(ws);
    __bf16* kb  = (__bf16*)(ws + tensor_bytes);
    __bf16* vT  = (__bf16*)(ws + 2 * tensor_bytes);
    __bf16* Wqb = (__bf16*)(ws + 3 * tensor_bytes);
    __bf16* Wkb = (__bf16*)(ws + 3 * tensor_bytes + w_bytes);
    float* out = (float*)d_out;

    const int wn = H * H;
    cvt_bf16_kernel<<<wn / (256 * 8), 256, 0, stream>>>(Wq, Wqb, wn);
    cvt_bf16_kernel<<<wn / (256 * 8), 256, 0, stream>>>(Wk, Wkb, wn);
    transpose_v_kernel<<<dim3(KK / 64, H / 64, BB), 256, 0, stream>>>(value, vT);
    proj_ln_kernel<<<(QQ * BB) / 32, 256, 0, stream>>>(query,  Wqb, bq, gq, betaq, qb, QQ);
    proj_ln_kernel<<<(KK * BB) / 32, 256, 0, stream>>>(key_in, Wkb, bk, gk, betak, kb, KK);
    attn_kernel<<<dim3(QQ / 32, BB), 256, 0, stream>>>(qb, kb, vT, qmask, kmask, out);
}
